// MinimalGazeEncoder_12670153523324
// MI455X (gfx1250) — matrix operand order repaired and checked by reading, not compiled, not run on hardware
//
#include <hip/hip_runtime.h>
#include <hip/hip_bf16.h>
#include <math.h>

// ---------- types ----------
typedef __attribute__((ext_vector_type(16))) __bf16 v16bf;
typedef __attribute__((ext_vector_type(8)))  float  v8f;
typedef __attribute__((ext_vector_type(4)))  float  f32x4;

union Frag16 { v16bf v; f32x4 q[2]; };

#define TT   8192
#define BBAT 64
#define MTOT (TT * BBAT)   // 524288 tokens
#define SEG  512           // EMA segment length: 0.95^512 ~ 4e-12 (carry negligible)

// Branch-free GELU. The WMMA path already quantizes to bf16 (~4e-3 rel),
// so the tanh-form deviation (<~3e-4) is below the accepted noise floor.
// Uses gfx1250 hardware v_tanh_f32 when the builtin exists.
__device__ __forceinline__ float gelu_fast(float x) {
    float u = x * x;
    float t = 0.7978845608028654f * x * (1.0f + 0.044715f * u);
#if __has_builtin(__builtin_amdgcn_tanhf)
    float th = __builtin_amdgcn_tanhf(t);
#else
    float e  = __expf(2.0f * t);          // t->+inf: e=inf -> th=1; t->-inf: e=0 -> th=-1
    float th = 1.0f - 2.0f / (e + 1.0f);
#endif
    return 0.5f * x * (1.0f + th);
}

// ---------------------------------------------------------------------------
// Kernel 1: segment-parallel causal EMA of the saccade gate.
// 64 rows x 16 segments of 512; gate recomputed inline from gaze (exact).
// Within a segment: wave32 associative scan of affine maps (a,b):
//   f(acc)=a*acc+b, compose (a1,b1)after(a2,b2) = (a1*a2, a1*b2+b1).
// Carry-in truncated at segment boundaries (error <= 0.95^512 ~ 4e-12).
// ---------------------------------------------------------------------------
__global__ __launch_bounds__(32) void ema_kernel(
    const float* __restrict__ gaze,
    const float* __restrict__ logthr, const float* __restrict__ invT,
    float* __restrict__ q)                 // q[m*2] = q_fast, q[m*2+1] = q_slow
{
    int b    = blockIdx.x >> 4;
    int s    = blockIdx.x & 15;
    int lane = threadIdx.x;
    float thr = expf(logthr[0]);
    float iT  = invT[0];
    size_t rb = (size_t)b * TT;
    float cf = 0.0f, cs = 0.0f;            // zero carry-in (see header comment)
    for (int c = 0; c < SEG / 32; c++) {
        int t  = s * SEG + c * 32 + lane;
        int t1 = t > 0 ? t - 1 : 0;
        const float* g0p = gaze + (rb + (size_t)t ) * 2;
        const float* g1p = gaze + (rb + (size_t)t1) * 2;
        float vx = (g0p[0] - g1p[0]) * 240.0f;
        float vy = (g0p[1] - g1p[1]) * 240.0f;
        float speed = sqrtf(vx * vx + vy * vy);
        float x = 1.0f / (1.0f + __expf(-iT * (speed - thr)));   // gate

        float Af = 0.8f,  Bf = 0.2f  * x;
        float As = 0.95f, Bs = 0.05f * x;
#pragma unroll
        for (int d = 1; d < 32; d <<= 1) {
            float Afo = __shfl_up(Af, d, 32), Bfo = __shfl_up(Bf, d, 32);
            float Aso = __shfl_up(As, d, 32), Bso = __shfl_up(Bs, d, 32);
            if (lane >= d) {
                Bf = Af * Bfo + Bf; Af = Af * Afo;
                Bs = As * Bso + Bs; As = As * Aso;
            }
        }
        float qf = Af * cf + Bf;
        float qs = As * cs + Bs;
        cf = __shfl(qf, 31, 32);
        cs = __shfl(qs, 31, 32);
        size_t m = rb + (size_t)t;
        q[m * 2]     = qf;
        q[m * 2 + 1] = qs;
    }
}

// ---------------------------------------------------------------------------
// Kernel 2: weight prep -> N-major bf16: w1t[128][32] (K padded), w2t[128][128]
// ---------------------------------------------------------------------------
__global__ __launch_bounds__(128) void prep_kernel(
    const float* __restrict__ W1, const float* __restrict__ W2,
    __bf16* __restrict__ w1t, __bf16* __restrict__ w2t)
{
    int n = threadIdx.x;   // output column
#pragma unroll
    for (int k = 0; k < 32; k++)
        w1t[n * 32 + k] = (k < 20) ? (__bf16)W1[k * 128 + n] : (__bf16)0.0f;
    for (int k = 0; k < 128; k++)
        w2t[n * 128 + k] = (__bf16)W2[k * 128 + n];
}

// ---------------------------------------------------------------------------
// Kernel 3: fused features -> GEMM1 -> gelu -> GEMM2 -> gelu.
// 128 threads = 4 waves; 64 tokens per WG; wave w owns rows [w*16, w*16+16).
// Features built straight into LDS (no global feats round-trip); only the
// EMA outputs come from the workspace. 40 v_wmma_f32_16x16x32_bf16 per wave.
// ---------------------------------------------------------------------------
__global__ __launch_bounds__(128) void gemm_kernel(
    const float*  __restrict__ gaze,
    const float*  __restrict__ logwx, const float* __restrict__ phix,
    const float*  __restrict__ logwy, const float* __restrict__ phiy,
    const float*  __restrict__ logthr, const float* __restrict__ invT,
    const float*  __restrict__ q,
    const __bf16* __restrict__ w1t, const __bf16* __restrict__ w2t,
    const float*  __restrict__ b1,  const float*  __restrict__ b2,
    float* __restrict__ out)
{
    __shared__ __bf16 sW1[128 * 32];    //  8 KB  (N-major)
    __shared__ __bf16 sW2[128 * 128];   // 32 KB  (N-major)
    __shared__ __bf16 sA [64  * 32];    //  4 KB  feature rows (K padded to 32)
    __shared__ __bf16 sH [64  * 128];   // 16 KB  hidden tile

    int tid = threadIdx.x;
    {   // coalesced staging of weights through L2
        const f32x4* s1 = (const f32x4*)w1t; f32x4* d1 = (f32x4*)sW1;
        for (int i = tid; i < 512;  i += 128) d1[i] = s1[i];
        const f32x4* s2 = (const f32x4*)w2t; f32x4* d2 = (f32x4*)sW2;
        for (int i = tid; i < 2048; i += 128) d2[i] = s2[i];
    }

    // ---- feature generation: threads 0..63 build one token row each ----
    if (tid < 64) {
        size_t m = (size_t)blockIdx.x * 64 + tid;
        int t = (int)(m & (TT - 1));
        size_t b = m >> 13;
        int t1 = t > 0 ? t - 1 : 0;       // clamped indices reproduce the
        int t2 = t > 1 ? t - 2 : 0;       // "prepend first frame" semantics
        size_t rb = b * (size_t)TT;
        const float* g0p = gaze + (rb + (size_t)t ) * 2;
        const float* g1p = gaze + (rb + (size_t)t1) * 2;
        const float* g2p = gaze + (rb + (size_t)t2) * 2;
        float gx = g0p[0], gy = g0p[1];
        float g1x = g1p[0], g1y = g1p[1];
        float g2x = g2p[0], g2y = g2p[1];

        const float rdt = 240.0f;
        float vx  = (gx  - g1x) * rdt, vy  = (gy  - g1y) * rdt;
        float vpx = (g1x - g2x) * rdt, vpy = (g1y - g2y) * rdt;
        float ax  = (vx - vpx) * rdt,  ay  = (vy - vpy) * rdt;
        float speed = sqrtf(vx * vx + vy * vy);
        float isp = 1.0f / (speed + 1e-6f);
        float dc = vx * isp, dsn = vy * isp;
        float apar  = (vx * ax + vy * ay) * isp;
        float aperp = (vx * ay - vy * ax) * isp;
        float thr  = expf(logthr[0]);
        float gate = 1.0f / (1.0f + __expf(-invT[0] * (speed - thr)));

        const float twopi = 6.28318530717958647692f;
        __bf16 row[32];
#pragma unroll
        for (int k = 0; k < 2; k++) {
            float w = expf(logwx[k]);
            float arg = twopi * gx * w + phix[k];
            row[k]     = (__bf16)sinf(arg);
            row[2 + k] = (__bf16)cosf(arg);
        }
#pragma unroll
        for (int k = 0; k < 2; k++) {
            float w = expf(logwy[k]);
            float arg = twopi * gy * w + phiy[k];
            row[4 + k] = (__bf16)sinf(arg);
            row[6 + k] = (__bf16)cosf(arg);
        }
        row[8]  = (__bf16)vx;    row[9]  = (__bf16)vy;   row[10] = (__bf16)speed;
        row[11] = (__bf16)dc;    row[12] = (__bf16)dsn;
        row[13] = (__bf16)ax;    row[14] = (__bf16)ay;
        row[15] = (__bf16)apar;  row[16] = (__bf16)aperp;
        row[17] = (__bf16)gate;
        row[18] = (__bf16)q[m * 2];       // q_fast from EMA pass
        row[19] = (__bf16)q[m * 2 + 1];   // q_slow from EMA pass
#pragma unroll
        for (int k = 20; k < 32; k++) row[k] = (__bf16)0.0f;

        f32x4* dst = (f32x4*)(sA + (size_t)tid * 32);
        const f32x4* src = (const f32x4*)row;
        dst[0] = src[0]; dst[1] = src[1]; dst[2] = src[2]; dst[3] = src[3];
    }
    __syncthreads();

    int wave = tid >> 5, lane = tid & 31;
    int lr = lane & 15, hi = lane >> 4;       // hi selects K-half per ISA A/B layout
    size_t r0 = (size_t)blockIdx.x * 64 + wave * 16;

    // ---- GEMM1: A = feature rows (16x32 bf16), one WMMA per 16x16 N-tile ----
    Frag16 a1;
    const __bf16* arow = sA + (wave * 16 + lr) * 32;
    a1.q[0] = *(const f32x4*)(arow + hi * 8);        // K = {0..7}   / {8..15}
    a1.q[1] = *(const f32x4*)(arow + 16 + hi * 8);   // K = {16..23} / {24..31}

    v8f acc1[8];
#pragma unroll
    for (int nt = 0; nt < 8; nt++) {
        Frag16 bfg;
        const __bf16* brow = sW1 + (nt * 16 + lr) * 32;
        bfg.q[0] = *(const f32x4*)(brow + hi * 8);
        bfg.q[1] = *(const f32x4*)(brow + 16 + hi * 8);
        v8f z = {0.f, 0.f, 0.f, 0.f, 0.f, 0.f, 0.f, 0.f};
        acc1[nt] = __builtin_amdgcn_wmma_f32_16x16x32_bf16(
            false, a1.v, false, bfg.v, (short)0, z, false, false);
    }

    // bias + gelu + write hidden tile to LDS as bf16
#pragma unroll
    for (int nt = 0; nt < 8; nt++) {
        int n = nt * 16 + lr;
        float bias = b1[n];
#pragma unroll
        for (int j = 0; j < 8; j++) {
            float x = acc1[nt][j] + bias;                     // D: M=j+8*hi, N=n
            sH[(wave * 16 + j + 8 * hi) * 128 + n] = (__bf16)gelu_fast(x);
        }
    }
    __syncthreads();

    // ---- GEMM2: K = 128 -> 4 WMMA K-steps x 8 N-tiles ----
    v8f acc2[8];
#pragma unroll
    for (int nt = 0; nt < 8; nt++) {
        v8f z = {0.f, 0.f, 0.f, 0.f, 0.f, 0.f, 0.f, 0.f};
        acc2[nt] = z;
    }
#pragma unroll
    for (int ks = 0; ks < 4; ks++) {
        Frag16 a2;
        const __bf16* hrow = sH + (wave * 16 + lr) * 128 + ks * 32;
        a2.q[0] = *(const f32x4*)(hrow + hi * 8);
        a2.q[1] = *(const f32x4*)(hrow + 16 + hi * 8);
#pragma unroll
        for (int nt = 0; nt < 8; nt++) {
            Frag16 bfg;
            const __bf16* brow = sW2 + (nt * 16 + lr) * 128 + ks * 32;
            bfg.q[0] = *(const f32x4*)(brow + hi * 8);
            bfg.q[1] = *(const f32x4*)(brow + 16 + hi * 8);
            acc2[nt] = __builtin_amdgcn_wmma_f32_16x16x32_bf16(
                false, a2.v, false, bfg.v, (short)0, acc2[nt], false, false);
        }
    }

    // bias + gelu + f32 output (store-bandwidth bound: 256 MB total)
#pragma unroll
    for (int nt = 0; nt < 8; nt++) {
        int n = nt * 16 + lr;
        float bias = b2[n];
#pragma unroll
        for (int j = 0; j < 8; j++) {
            float x = acc2[nt][j] + bias;
            out[(r0 + (size_t)(j + 8 * hi)) * 128 + n] = gelu_fast(x);
        }
    }
}

// ---------------------------------------------------------------------------
extern "C" void kernel_launch(void* const* d_in, const int* in_sizes, int n_in,
                              void* d_out, int out_size, void* d_ws, size_t ws_size,
                              hipStream_t stream)
{
    const float* gaze   = (const float*)d_in[0];
    const float* logwx  = (const float*)d_in[1];
    const float* phix   = (const float*)d_in[2];
    const float* logwy  = (const float*)d_in[3];
    const float* phiy   = (const float*)d_in[4];
    const float* logthr = (const float*)d_in[5];
    const float* invT   = (const float*)d_in[6];
    const float* W1     = (const float*)d_in[7];
    const float* b1     = (const float*)d_in[8];
    const float* W2     = (const float*)d_in[9];
    const float* b2     = (const float*)d_in[10];

    char* ws = (char*)d_ws;
    float*  q   = (float*)ws;                                  // 4 MB (qf,qs pairs)
    __bf16* w1t = (__bf16*)(ws + (size_t)MTOT * 8);            // 8 KB
    __bf16* w2t = w1t + 128 * 32;                              // 32 KB
    float*  out = (float*)d_out;

    ema_kernel<<<BBAT * (TT / SEG), 32, 0, stream>>>(gaze, logthr, invT, q);
    prep_kernel<<<1, 128, 0, stream>>>(W1, W2, w1t, w2t);
    gemm_kernel<<<MTOT / 64, 128, 0, stream>>>(gaze, logwx, phix, logwy, phiy,
                                               logthr, invT, q, w1t, w2t, b1, b2, out);
}
